// Gemma3ForMultimodalLMwithTPA_13288628814268
// MI455X (gfx1250) — matrix operand order repaired and checked by reading, not compiled, not run on hardware
//
#include <hip/hip_runtime.h>

#define Bb   4
#define Ss   2048
#define HIDd 2048
#define NHh  16
#define NKVv 8
#define HDd  128
#define QRr  6
#define KRr  2
#define VRr  2
#define PCOLS 1408
#define EPSf 1e-6f
#define SCALEf 0.08838834764831845f   /* 128^-0.5 */
#define MASK_NEGf -2.3819763e38f

typedef __attribute__((ext_vector_type(16))) __bf16 bf16x16;
typedef __attribute__((ext_vector_type(8)))  float  f32x8;
typedef __attribute__((ext_vector_type(8)))  unsigned int u32x8;
typedef __attribute__((ext_vector_type(4)))  unsigned int u32x4;
typedef __attribute__((ext_vector_type(8)))  int i32x8;
typedef __attribute__((ext_vector_type(4)))  int i32x4;

union FragB { u32x8 u; bf16x16 b; };

static __device__ __forceinline__ unsigned short f2bf(float f) {
  unsigned int u = __float_as_uint(f);
  unsigned int r = (u + 0x7FFFu + ((u >> 16) & 1u)) >> 16;
  return (unsigned short)r;
}

// ---- TDM: async 2D tile load global -> LDS (D# per CDNA5 ISA) ----
// tile_d0 elements contiguous (data_size=2B), tile_d1 rows, row stride stride0 elements.
// LDS padding: pad_amount DWORDs inserted every pad_interval DWORDs (codes per ISA table).
static __device__ __forceinline__ void tdm_load_tile_2d(
    unsigned int lds_addr, const unsigned short* gptr,
    unsigned int tile_d0, unsigned int tile_d1,
    unsigned int tdim0, unsigned int tdim1, unsigned int stride0,
    unsigned int pad_interval_code, unsigned int pad_amount_code)
{
  unsigned long long ga = (unsigned long long)(size_t)gptr;
  u32x4 g0;
  g0[0] = 1u;                                             // count=1, user mode
  g0[1] = lds_addr;                                       // LDS byte address
  g0[2] = (unsigned int)ga;                               // global_addr[31:0]
  g0[3] = (unsigned int)((ga >> 32) & 0x1FFFFFFu) | (2u << 30); // [56:32] | type=2
  i32x8 g1;
  g1[0] = (int)((1u << 16) |                              // data_size = 1 -> 2 bytes
                (1u << 20) |                              // pad_enable
                (pad_interval_code << 22) |
                (pad_amount_code << 25));
  g1[1] = (int)((tdim0 & 0xFFFFu) << 16);                 // tensor_dim0[15:0] @ bit48
  g1[2] = (int)((tdim0 >> 16) | ((tdim1 & 0xFFFFu) << 16)); // dim0[31:16] | dim1[15:0]
  g1[3] = (int)((tdim1 >> 16) | (tile_d0 << 16));         // dim1[31:16] | tile_dim0
  g1[4] = (int)(tile_d1);                                 // tile_dim1 | tile_dim2=0
  g1[5] = (int)stride0;                                   // tensor_dim0_stride[31:0]
  g1[6] = 0;                                              // stride0[47:32]|stride1[15:0]
  g1[7] = 0;
  i32x4 z4 = {0, 0, 0, 0};
  i32x8 z8 = {0, 0, 0, 0, 0, 0, 0, 0};
  __builtin_amdgcn_tensor_load_to_lds(g0, g1, z4, z4, z8, 0);
}

static __device__ __forceinline__ unsigned int lds_off(const void* p) {
  return (unsigned int)(size_t)p;   // generic addr of __shared__ truncates to LDS offset
}

// ---------------- elementwise f32 -> bf16 ----------------
__global__ void f32_to_bf16_kernel(const float* __restrict__ src,
                                   unsigned short* __restrict__ dst, int n) {
  int i = blockIdx.x * blockDim.x + threadIdx.x;
  if (i < n) dst[i] = f2bf(src[i]);
}

// transpose (R x Cc f32) -> (Cc x R bf16)
__global__ void transpose_f32_bf16_kernel(const float* __restrict__ src,
                                          unsigned short* __restrict__ dst,
                                          int R, int Cc, int n) {
  int i = blockIdx.x * blockDim.x + threadIdx.x;
  if (i < n) {
    int r = i / Cc, c = i - r * Cc;
    dst[(size_t)c * R + r] = f2bf(src[i]);
  }
}

// pack one projection weight (HID x nc, f32) TRANSPOSED into WcatT (PCOLS x HID, bf16)
__global__ void pack_colsT_kernel(const float* __restrict__ src,
                                  unsigned short* __restrict__ dst,
                                  int nc, int coloff, int n) {
  int i = blockIdx.x * blockDim.x + threadIdx.x;
  if (i < n) {
    int row = i / nc;
    int c = i - row * nc;
    dst[(size_t)(coloff + c) * HIDd + row] = f2bf(src[i]);
  }
}

// ---------------- bf16 WMMA GEMM with TDM double buffering ----------------
// C[f32, MxN] = A[bf16, MxK row-major] * BT[bf16, NxK row-major]^T
// Tiles 128x128x32; 8 waves (4x2), each wave 32x64 = 2x4 wmma tiles.
// Both LDS tiles are [128][40] ushort (hardware-padded rows: 64B data + 16B pad).
__global__ __launch_bounds__(256) void gemm_bf16_kernel(
    const unsigned short* __restrict__ A, const unsigned short* __restrict__ BT,
    float* __restrict__ C, int M, int N, int K)
{
  __shared__ unsigned short sA[2][128][40];
  __shared__ unsigned short sB[2][128][40];
  const int tid  = threadIdx.x;
  const int lane = tid & 31;
  const int wave = tid >> 5;
  const int wr = wave >> 1, wc = wave & 1;
  const int ln = lane & 15, lh = lane >> 4;
  const int rowBase = blockIdx.x * 128;
  const int colBase = blockIdx.y * 128;
  const int KT = K >> 5;

  f32x8 acc[2][4];
#pragma unroll
  for (int i = 0; i < 2; ++i)
#pragma unroll
    for (int j = 0; j < 4; ++j) acc[i][j] = (f32x8){0,0,0,0,0,0,0,0};

  if (wave == 0) {   // prologue: async-load tile 0 into buffer 0
    tdm_load_tile_2d(lds_off(&sA[0][0][0]), A  + (size_t)rowBase * K, 32, 128, K, M, K, 3, 3);
    tdm_load_tile_2d(lds_off(&sB[0][0][0]), BT + (size_t)colBase * K, 32, 128, K, N, K, 3, 3);
  }

  for (int kt = 0; kt < KT; ++kt) {
    const int cur = kt & 1;
    if (wave == 0) {
      if (kt + 1 < KT) {       // prefetch next tile into the other buffer
        const int k0n = (kt + 1) << 5;
        tdm_load_tile_2d(lds_off(&sA[cur ^ 1][0][0]), A  + (size_t)rowBase * K + k0n,
                         32, 128, K, M, K, 3, 3);
        tdm_load_tile_2d(lds_off(&sB[cur ^ 1][0][0]), BT + (size_t)colBase * K + k0n,
                         32, 128, K, N, K, 3, 3);
        __builtin_amdgcn_s_wait_tensorcnt(2);   // current tile's pair complete
      } else {
        __builtin_amdgcn_s_wait_tensorcnt(0);
      }
    }
    __syncthreads();

    FragB af[2];
#pragma unroll
    for (int mr = 0; mr < 2; ++mr) {       // A frag 16x32
      int row = wr * 32 + mr * 16 + ln;
      int kh = lh * 8;
#pragma unroll
      for (int i = 0; i < 4; ++i) {
        af[mr].u[i]     = *(const unsigned int*)&sA[cur][row][kh + 2 * i];
        af[mr].u[4 + i] = *(const unsigned int*)&sA[cur][row][16 + kh + 2 * i];
      }
    }
#pragma unroll
    for (int nc = 0; nc < 4; ++nc) {
      FragB bfrag;                          // B frag 32x16 from row of BT
      int col = wc * 64 + nc * 16 + ln;
      int kb = lh * 16;
#pragma unroll
      for (int i = 0; i < 8; ++i)
        bfrag.u[i] = *(const unsigned int*)&sB[cur][col][(kb >> 1) + (i >> 2) * 16 + 2 * (i & 3)];
#pragma unroll
      for (int mr = 0; mr < 2; ++mr)
        acc[mr][nc] = __builtin_amdgcn_wmma_f32_16x16x32_bf16(
            false, af[mr].b, false, bfrag.b, (short)0, acc[mr][nc], false, false);
    }
    __syncthreads();   // everyone done reading buf[cur] before it is refilled
  }
#pragma unroll
  for (int mr = 0; mr < 2; ++mr)
#pragma unroll
    for (int nc = 0; nc < 4; ++nc) {
      int col = colBase + wc * 64 + nc * 16 + ln;
#pragma unroll
      for (int r = 0; r < 8; ++r) {
        int row = rowBase + wr * 32 + mr * 16 + r + lh * 8;
        C[(size_t)row * N + col] = acc[mr][nc][r];
      }
    }
}

// ---------------- per-token: RoPE + rank contraction + RMS-norm -> Q/K/V(bf16) ----------------
__global__ __launch_bounds__(128) void prep_kernel(
    const float* __restrict__ P, const float* __restrict__ cosT, const float* __restrict__ sinT,
    const float* __restrict__ qnw, const float* __restrict__ knw,
    unsigned short* __restrict__ Qb, unsigned short* __restrict__ Kb,
    unsigned short* __restrict__ VT)
{
  const int token = blockIdx.x;
  const int b = token / Ss;
  const int s = token - b * Ss;
  const int d = threadIdx.x;
  const float* row = P + (size_t)token * PCOLS;
  __shared__ float sA[128];          // first 128 cols = A_q(96) | A_k(16) | A_v(16)
  __shared__ float sred[16 * 4];
  sA[d] = row[d];
  __syncthreads();

  const int dd = d & 63;
  const bool hi = d >= 64;
  const float cv = cosT[s * 64 + dd];
  const float sv = sinT[s * 64 + dd];

  float bq[QRr];
#pragma unroll
  for (int r = 0; r < QRr; ++r) {
    const float* base = row + 128 + r * HDd;
    float xr = base[dd], xi = base[64 + dd];
    bq[r] = hi ? (xr * sv + xi * cv) : (xr * cv - xi * sv);
  }
  float bk[KRr];
#pragma unroll
  for (int r = 0; r < KRr; ++r) {
    const float* base = row + 896 + r * HDd;
    float xr = base[dd], xi = base[64 + dd];
    bk[r] = hi ? (xr * sv + xi * cv) : (xr * cv - xi * sv);
  }
  float bv[VRr];
#pragma unroll
  for (int r = 0; r < VRr; ++r) bv[r] = row[1152 + r * HDd + d];

  const int lane = d & 31, wid = d >> 5;

  float qh[NHh];
#pragma unroll
  for (int h = 0; h < NHh; ++h) {
    float a = 0.f;
#pragma unroll
    for (int r = 0; r < QRr; ++r) a += sA[h * QRr + r] * bq[r];
    qh[h] = a * (1.f / QRr);
    float v2 = qh[h] * qh[h];
#pragma unroll
    for (int off = 16; off; off >>= 1) v2 += __shfl_xor(v2, off, 32);
    if (lane == 0) sred[h * 4 + wid] = v2;
  }
  __syncthreads();
  const float qw = 1.f + qnw[d];
#pragma unroll
  for (int h = 0; h < NHh; ++h) {
    float sum = sred[h * 4] + sred[h * 4 + 1] + sred[h * 4 + 2] + sred[h * 4 + 3];
    float rn = rsqrtf(sum * (1.f / HDd) + EPSf);
    Qb[(((size_t)(b * NHh + h)) * Ss + s) * HDd + d] = f2bf(qh[h] * rn * qw * SCALEf);
  }
  __syncthreads();

  float kg[NKVv];
#pragma unroll
  for (int g = 0; g < NKVv; ++g) {
    float a = sA[96 + g * KRr] * bk[0] + sA[96 + g * KRr + 1] * bk[1];
    kg[g] = a * 0.5f;
    float v2 = kg[g] * kg[g];
#pragma unroll
    for (int off = 16; off; off >>= 1) v2 += __shfl_xor(v2, off, 32);
    if (lane == 0) sred[g * 4 + wid] = v2;
  }
  __syncthreads();
  const float kw = 1.f + knw[d];
#pragma unroll
  for (int g = 0; g < NKVv; ++g) {
    float sum = sred[g * 4] + sred[g * 4 + 1] + sred[g * 4 + 2] + sred[g * 4 + 3];
    float rn = rsqrtf(sum * (1.f / HDd) + EPSf);
    Kb[(((size_t)(b * NKVv + g)) * Ss + s) * HDd + d] = f2bf(kg[g] * rn * kw);
    float v = (sA[112 + g * VRr] * bv[0] + sA[112 + g * VRr + 1] * bv[1]) * 0.5f;
    VT[(((size_t)(b * NKVv + g)) * HDd + d) * Ss + s] = f2bf(v);
  }
}

// ---------------- FlashAttention (causal, GQA rep=2), WMMA bf16 + TDM KV staging ----------------
// Block = 4 waves, each wave a 16-row Q strip; KV tiles of 32 double-buffered in LDS via TDM.
__global__ __launch_bounds__(128) void attn_kernel(
    const unsigned short* __restrict__ Qb, const unsigned short* __restrict__ Kb,
    const unsigned short* __restrict__ VT, unsigned short* __restrict__ Ob)
{
  const int qt = blockIdx.x;      // 0..31 (64-row q blocks)
  const int h  = blockIdx.y;      // 0..15
  const int b  = blockIdx.z;      // 0..3
  const int g  = h >> 1;          // kv head (rep = 2)
  const int lane = threadIdx.x & 31;
  const int wave = threadIdx.x >> 5;
  const int ln = lane & 15, lh = lane >> 4;
  const int q0 = qt * 64 + wave * 16;

  __shared__ unsigned short sK[2][32][136];  // K tile: 32 kv rows x 128 d (+16B row pad)
  __shared__ unsigned short sV[2][128][40];  // V^T tile: 128 d rows x 32 kv (+16B row pad)
  __shared__ unsigned short sP[4][16][32];   // per-wave probs staging (C->A layout)

  const unsigned short* kbase = Kb + ((size_t)(b * NKVv + g)) * Ss * HDd;
  const unsigned short* vbase = VT + ((size_t)(b * NKVv + g)) * HDd * Ss;

  FragB qf[4];                               // 16x128 Q strip as 4 A-frags (16x32)
  {
    const unsigned short* qbase =
        Qb + ((size_t)(b * NHh + h)) * Ss * HDd + (size_t)(q0 + ln) * HDd;
    int kh = lh * 8;
#pragma unroll
    for (int c = 0; c < 4; ++c) {
      const unsigned short* p = qbase + c * 32;
#pragma unroll
      for (int i = 0; i < 4; ++i) {
        qf[c].u[i]     = *(const unsigned int*)(p + kh + 2 * i);
        qf[c].u[4 + i] = *(const unsigned int*)(p + 16 + kh + 2 * i);
      }
    }
  }
  f32x8 acc[8];
#pragma unroll
  for (int i = 0; i < 8; ++i) acc[i] = (f32x8){0,0,0,0,0,0,0,0};
  float m[8], l[8];
#pragma unroll
  for (int r = 0; r < 8; ++r) { m[r] = -3.0e38f; l[r] = 0.f; }

  const int KT = qt * 2 + 2;      // number of 32-wide kv tiles (causal bound)

  if (wave == 0) {   // prologue: K tile rows contiguous in d; V^T tile rows contiguous in s
    tdm_load_tile_2d(lds_off(&sK[0][0][0]), kbase, 128, 32, HDd, Ss, HDd, 5, 3);
    tdm_load_tile_2d(lds_off(&sV[0][0][0]), vbase, 32, 128, Ss, HDd, Ss, 3, 3);
  }

  for (int kt = 0; kt < KT; ++kt) {
    const int cur = kt & 1;
    const int k0 = kt << 5;
    if (wave == 0) {
      if (kt + 1 < KT) {
        const int k0n = (kt + 1) << 5;
        tdm_load_tile_2d(lds_off(&sK[cur ^ 1][0][0]), kbase + (size_t)k0n * HDd,
                         128, 32, HDd, Ss, HDd, 5, 3);
        tdm_load_tile_2d(lds_off(&sV[cur ^ 1][0][0]), vbase + k0n,
                         32, 128, Ss, HDd, Ss, 3, 3);
        __builtin_amdgcn_s_wait_tensorcnt(2);
      } else {
        __builtin_amdgcn_s_wait_tensorcnt(0);
      }
    }
    __syncthreads();

    f32x8 sc[2];
    sc[0] = (f32x8){0,0,0,0,0,0,0,0};
    sc[1] = (f32x8){0,0,0,0,0,0,0,0};
#pragma unroll
    for (int j = 0; j < 2; ++j) {          // S = Q * K^T (two 16-col tiles)
#pragma unroll
      for (int c = 0; c < 4; ++c) {
        FragB kf;
#pragma unroll
        for (int i = 0; i < 8; ++i)
          kf.u[i] = *(const unsigned int*)&sK[cur][j * 16 + ln][c * 32 + lh * 8 + (i >> 2) * 16 + 2 * (i & 3)];
        sc[j] = __builtin_amdgcn_wmma_f32_16x16x32_bf16(
            false, qf[c].b, false, kf.b, (short)0, sc[j], false, false);
      }
    }
    // online softmax: rows live in 16-lane halves -> shfl_xor 1/2/4/8 reduces one row
#pragma unroll
    for (int r = 0; r < 8; ++r) {
      int rowq = q0 + r + lh * 8;
      float s0 = (k0 + ln <= rowq)      ? sc[0][r] : MASK_NEGf;
      float s1 = (k0 + 16 + ln <= rowq) ? sc[1][r] : MASK_NEGf;
      float mx = fmaxf(s0, s1);
#pragma unroll
      for (int off = 8; off; off >>= 1) mx = fmaxf(mx, __shfl_xor(mx, off, 32));
      float mn = fmaxf(m[r], mx);
      float scl = __expf(m[r] - mn);
      float p0 = __expf(s0 - mn);
      float p1 = __expf(s1 - mn);
      float rs = p0 + p1;
#pragma unroll
      for (int off = 8; off; off >>= 1) rs += __shfl_xor(rs, off, 32);
      l[r] = l[r] * scl + rs;
      m[r] = mn;
#pragma unroll
      for (int dt = 0; dt < 8; ++dt) acc[dt][r] *= scl;
      sP[wave][r + lh * 8][ln]      = f2bf(p0);
      sP[wave][r + lh * 8][16 + ln] = f2bf(p1);
    }
    __syncthreads();
    FragB pf;                               // re-read probs in A-layout
    {
      int kh = lh * 8;
#pragma unroll
      for (int i = 0; i < 4; ++i) {
        pf.u[i]     = *(const unsigned int*)&sP[wave][ln][kh + 2 * i];
        pf.u[4 + i] = *(const unsigned int*)&sP[wave][ln][16 + kh + 2 * i];
      }
    }
#pragma unroll
    for (int dt = 0; dt < 8; ++dt) {        // O += P * V
      FragB vf;
#pragma unroll
      for (int i = 0; i < 8; ++i)
        vf.u[i] = *(const unsigned int*)&sV[cur][dt * 16 + ln][lh * 8 + (i >> 2) * 16 + 2 * (i & 3)];
      acc[dt] = __builtin_amdgcn_wmma_f32_16x16x32_bf16(
          false, pf.b, false, vf.b, (short)0, acc[dt], false, false);
    }
    __syncthreads();   // everyone done with buf[cur] before refill
  }
#pragma unroll
  for (int r = 0; r < 8; ++r) {
    float inv = 1.f / l[r];
    size_t rowoff = ((size_t)b * Ss + (q0 + r + lh * 8)) * (NHh * HDd) + h * HDd;
#pragma unroll
    for (int dt = 0; dt < 8; ++dt)
      Ob[rowoff + dt * 16 + ln] = f2bf(acc[dt][r] * inv);
  }
}

// ---------------- host launch ----------------
extern "C" void kernel_launch(void* const* d_in, const int* in_sizes, int n_in,
                              void* d_out, int out_size, void* d_ws, size_t ws_size,
                              hipStream_t stream) {
  (void)in_sizes; (void)n_in; (void)out_size; (void)ws_size;
  const float* hsrc = (const float*)d_in[0];
  const float* fc   = (const float*)d_in[1];
  const float* fs   = (const float*)d_in[2];
  // d_in[3] mask, d_in[4] kv_write_indices: causality handled analytically
  const float* waq = (const float*)d_in[5];
  const float* wak = (const float*)d_in[6];
  const float* wav = (const float*)d_in[7];
  const float* wbq = (const float*)d_in[8];
  const float* wbk = (const float*)d_in[9];
  const float* wbv = (const float*)d_in[10];
  const float* wo  = (const float*)d_in[11];
  const float* qnw = (const float*)d_in[12];
  const float* knw = (const float*)d_in[13];
  float* out = (float*)d_out;

  char* ws = (char*)d_ws;
  size_t off = 0;
  auto alloc = [&](size_t bytes) -> char* {
    char* p = ws + off;
    off += (bytes + 255) & ~(size_t)255;
    return p;
  };
  unsigned short* Xbf   = (unsigned short*)alloc((size_t)8192 * 2048 * 2);
  unsigned short* WcatT = (unsigned short*)alloc((size_t)1408 * 2048 * 2); // (PCOLS x HID)
  unsigned short* WobT  = (unsigned short*)alloc((size_t)2048 * 2048 * 2); // (N x K)
  float*          Pf    = (float*)alloc((size_t)8192 * 1408 * 4);
  unsigned short* Qb    = (unsigned short*)alloc((size_t)4 * 16 * 2048 * 128 * 2);
  unsigned short* Kb    = (unsigned short*)alloc((size_t)4 * 8 * 2048 * 128 * 2);
  unsigned short* VT    = (unsigned short*)alloc((size_t)4 * 8 * 2048 * 128 * 2);
  unsigned short* Attn  = (unsigned short*)alloc((size_t)8192 * 2048 * 2);

  { int n = 8192 * 2048;
    f32_to_bf16_kernel<<<(n + 255) / 256, 256, 0, stream>>>(hsrc, Xbf, n); }
  { int n = 2048 * 2048;
    transpose_f32_bf16_kernel<<<(n + 255) / 256, 256, 0, stream>>>(wo, WobT, 2048, 2048, n); }
  { int n = 2048 * 96;
    pack_colsT_kernel<<<(n + 255) / 256, 256, 0, stream>>>(waq, WcatT, 96, 0, n); }
  { int n = 2048 * 16;
    pack_colsT_kernel<<<(n + 255) / 256, 256, 0, stream>>>(wak, WcatT, 16, 96, n); }
  { int n = 2048 * 16;
    pack_colsT_kernel<<<(n + 255) / 256, 256, 0, stream>>>(wav, WcatT, 16, 112, n); }
  { int n = 2048 * 768;
    pack_colsT_kernel<<<(n + 255) / 256, 256, 0, stream>>>(wbq, WcatT, 768, 128, n); }
  { int n = 2048 * 256;
    pack_colsT_kernel<<<(n + 255) / 256, 256, 0, stream>>>(wbk, WcatT, 256, 896, n); }
  { int n = 2048 * 256;
    pack_colsT_kernel<<<(n + 255) / 256, 256, 0, stream>>>(wbv, WcatT, 256, 1152, n); }

  // fused projection GEMM: (8192 x 2048) x (2048 x 1408) -> P (f32)
  gemm_bf16_kernel<<<dim3(64, 11), 256, 0, stream>>>(Xbf, WcatT, Pf, 8192, 1408, 2048);
  // RoPE + rank contraction + RMS-norm -> bf16 Q/K/V^T
  prep_kernel<<<8192, 128, 0, stream>>>(Pf, fc, fs, qnw, knw, Qb, Kb, VT);
  // causal flash attention
  attn_kernel<<<dim3(32, 16, 4), 128, 0, stream>>>(Qb, Kb, VT, Attn);
  // output projection: (8192 x 2048) x (2048 x 2048) -> out (f32)
  gemm_bf16_kernel<<<dim3(64, 16), 256, 0, stream>>>(Attn, WobT, out, 8192, 2048, 2048);
}
